// MultiChannelLSTM_23957327577539
// MI455X (gfx1250) — matrix operand order repaired and checked by reading, not compiled, not run on hardware
//
#include <hip/hip_runtime.h>
#include <hip/hip_bf16.h>
#include <stdint.h>

#define BATCH 256
#define NCH   6
#define TLEN  2048
#define HID   16

typedef __attribute__((ext_vector_type(8)))  float  v8f;
typedef __attribute__((ext_vector_type(2)))  float  v2f;
typedef __attribute__((ext_vector_type(16))) __bf16 v16bf;

__device__ __forceinline__ float fast_sig(float x) {
  return __builtin_amdgcn_rcpf(1.0f + __expf(-x));
}
__device__ __forceinline__ float fast_tanh(float x) {
  return fmaf(2.0f, __builtin_amdgcn_rcpf(1.0f + __expf(-2.0f * x)), -1.0f);
}
__device__ __forceinline__ unsigned short f2bf(float x) {
  unsigned u = __float_as_uint(x);
  u += 0x7fffu + ((u >> 16) & 1u);       // round-to-nearest-even
  return (unsigned short)(u >> 16);
}

// ---------------------------------------------------------------------------
// Layer 0: bidirectional LSTM, input size 1. One WG per (ch, dir, mtile).
// 4 waves, wave w owns gate w (0=i,1=f,2=g,3=o).
// Per step: C=bias splat; 4x wmma_f32_16x16x4 (h@Whh^T) + 1x wmma (x*wih term
// folded as K-chunk 16..19). Writes h tile to global y as bf16.
// ---------------------------------------------------------------------------
__global__ __launch_bounds__(128)
void lstm_l0(const float* __restrict__ x,
             const float* __restrict__ wih,  const float* __restrict__ whh,
             const float* __restrict__ bih,  const float* __restrict__ bhh,
             unsigned* __restrict__ y /* packed bf16 pairs */) {
  const int tid  = threadIdx.x;
  const int w    = tid >> 5;          // gate index
  const int lane = tid & 31;
  const int n    = lane & 15;         // N column / A row / batch-in-tile
  const int hi   = lane >> 4;

  const int bx  = blockIdx.x;
  const int mt  = bx & 15;            // batch tile
  const int dir = (bx >> 4) & 1;
  const int ch  = bx >> 5;
  const int cd  = ch * 2 + dir;

  // Whh B-fragments (f32 16x16x4 layout): bB[2q+r] = Whh[gate*16+n][4q+r+2hi]
  const float* whhp = whh + ((size_t)cd * 64 + w * 16 + n) * 16;
  float bB[8];
#pragma unroll
  for (int q = 0; q < 4; q++) {
    bB[2 * q]     = whhp[4 * q +     2 * hi];
    bB[2 * q + 1] = whhp[4 * q + 1 + 2 * hi];
  }
  const float wn  = wih[(size_t)cd * 64 + w * 16 + n];
  const float bn  = bih[(size_t)cd * 64 + w * 16 + n] +
                    bhh[(size_t)cd * 64 + w * 16 + n];
  const float xB0 = hi ? 0.0f : wn;   // B chunk: row K=16 carries wih, rest 0

  __shared__ float gbuf[2][4][16 * 18];   // double-buffered gate tiles

  const float* xp = x + (size_t)(mt * 16) * NCH * TLEN + (size_t)ch * TLEN;
  unsigned* yb = y + (size_t)(ch * 16 + mt) * TLEN * 256;  // 256 u32 per t

  float cA[8], hA[8];
#pragma unroll
  for (int i = 0; i < 8; i++) { cA[i] = 0.0f; hA[i] = 0.0f; }

  const int tt0 = dir ? (TLEN - 1) : 0;
  float xv = (lane < 16) ? xp[(size_t)lane * NCH * TLEN + tt0] : 0.0f;

  for (int t = 0; t < TLEN; ++t) {
    const int p  = t & 1;
    const int tt = dir ? (TLEN - 1 - t) : t;
    const int tn  = (t + 1 < TLEN) ? (t + 1) : t;
    const int ttn = dir ? (TLEN - 1 - tn) : tn;
    float xv_next = (lane < 16) ? xp[(size_t)lane * NCH * TLEN + ttn] : 0.0f;

    // gates = bias + h @ Whh^T + x * wih
    v8f acc;
#pragma unroll
    for (int i = 0; i < 8; i++) acc[i] = bn;
#pragma unroll
    for (int q = 0; q < 4; q++) {
      v2f av = { hA[2 * q], hA[2 * q + 1] };
      v2f bv = { bB[2 * q], bB[2 * q + 1] };
      acc = __builtin_amdgcn_wmma_f32_16x16x4_f32(false, av, false, bv,
                                                  (short)0, acc, false, false);
    }
    {
      v2f av = { xv, 0.0f };
      v2f bv = { xB0, 0.0f };
      acc = __builtin_amdgcn_wmma_f32_16x16x4_f32(false, av, false, bv,
                                                  (short)0, acc, false, false);
    }

    // activate own gate tile in D layout
    float act[8];
    if (w == 2) {
#pragma unroll
      for (int i = 0; i < 8; i++) act[i] = fast_tanh(acc[i]);
    } else {
#pragma unroll
      for (int i = 0; i < 8; i++) act[i] = fast_sig(acc[i]);
    }
    float* gp = &gbuf[p][w][0];
#pragma unroll
    for (int v = 0; v < 8; v++) gp[(v + 8 * hi) * 18 + n] = act[v];
    __syncthreads();

    // combine in A-layout coordinates (row m=n, col k=4q+r+2hi)
    const float* g0 = &gbuf[p][0][0];
    const float* g1 = &gbuf[p][1][0];
    const float* g2 = &gbuf[p][2][0];
    const float* g3 = &gbuf[p][3][0];
#pragma unroll
    for (int q = 0; q < 4; q++) {
#pragma unroll
      for (int r = 0; r < 2; r++) {
        const int off = n * 18 + 4 * q + r + 2 * hi;
        float c = g1[off] * cA[2 * q + r] + g0[off] * g2[off];
        cA[2 * q + r] = c;
        hA[2 * q + r] = g3[off] * fast_tanh(c);
      }
    }

    // wave 0 stores h tile to y (bf16 pairs): feature f = dir*16 + k
    if (w == 0) {
      unsigned* yt = yb + (size_t)tt * 256;
#pragma unroll
      for (int q = 0; q < 4; q++) {
        unsigned pk = (unsigned)f2bf(hA[2 * q]) |
                      ((unsigned)f2bf(hA[2 * q + 1]) << 16);
        yt[n * 16 + dir * 8 + 2 * q + hi] = pk;
      }
    }
    xv = xv_next;
  }
}

// ---------------------------------------------------------------------------
// Layer 1: input = 32 bf16 features (both layer-0 directions).
// Per step: 1x v_wmma_f32_16x16x32_bf16 (y@Wih1^T, K=32 exact) +
//           4x v_wmma_f32_16x16x4_f32 (h@Whh1^T). Only final h written out.
// ---------------------------------------------------------------------------
__global__ __launch_bounds__(128)
void lstm_l1(const unsigned* __restrict__ y,
             const float* __restrict__ wih1, const float* __restrict__ whh1,
             const float* __restrict__ bih1, const float* __restrict__ bhh1,
             float* __restrict__ out) {
  const int tid  = threadIdx.x;
  const int w    = tid >> 5;
  const int lane = tid & 31;
  const int n    = lane & 15;
  const int hi   = lane >> 4;

  const int bx  = blockIdx.x;
  const int mt  = bx & 15;
  const int dir = (bx >> 4) & 1;
  const int ch  = bx >> 5;
  const int cd  = ch * 2 + dir;

  // Whh1 B-fragments (f32)
  const float* whhp = whh1 + ((size_t)cd * 64 + w * 16 + n) * 16;
  float bB[8];
#pragma unroll
  for (int q = 0; q < 4; q++) {
    bB[2 * q]     = whhp[4 * q +     2 * hi];
    bB[2 * q + 1] = whhp[4 * q + 1 + 2 * hi];
  }
  // Wih1 B-fragments (bf16 32x16): reg j -> K = {2j,2j+1} + 8*hi (+8 for j>=4), col n
  const float* wip = wih1 + ((size_t)cd * 64 + w * 16 + n) * 32;
  union { unsigned u[8]; v16bf v; } wB;
#pragma unroll
  for (int j = 0; j < 8; j++) {
    const int k0 = 2 * j + 8 * hi + 8 * (j >> 2);
    wB.u[j] = (unsigned)f2bf(wip[k0]) | ((unsigned)f2bf(wip[k0 + 1]) << 16);
  }
  const float bn = bih1[(size_t)cd * 64 + w * 16 + n] +
                   bhh1[(size_t)cd * 64 + w * 16 + n];

  __shared__ float gbuf[2][4][16 * 18];
  __shared__ uint2 ybuf[2][128];          // 1KB double-buffered y_t tile

  const uint2* yq = (const uint2*)y + (size_t)(ch * 16 + mt) * TLEN * 128;

  float cA[8], hA[8];
#pragma unroll
  for (int i = 0; i < 8; i++) { cA[i] = 0.0f; hA[i] = 0.0f; }

  // stage t = 0 tile
  {
    const int tt0 = dir ? (TLEN - 1) : 0;
    ybuf[0][tid] = yq[(size_t)tt0 * 128 + tid];
  }
  __syncthreads();

  for (int t = 0; t < TLEN; ++t) {
    const int p = t & 1;
    const int tn  = (t + 1 < TLEN) ? (t + 1) : t;
    const int ttn = dir ? (TLEN - 1 - tn) : tn;
    uint2 stage = yq[(size_t)ttn * 128 + tid];        // prefetch next tile

    // read y_t A-fragments (bf16, A layout) from ybuf[p]
    const uint4* yb4 = (const uint4*)&ybuf[p][0];
    union { uint4 q[2]; v16bf v; } Ay;
    Ay.q[0] = yb4[n * 4 + hi];        // K = 0..7  (+8 in hi half)
    Ay.q[1] = yb4[n * 4 + 2 + hi];    // K = 16..23 (+8 in hi half)

    v8f acc;
#pragma unroll
    for (int i = 0; i < 8; i++) acc[i] = bn;
    acc = __builtin_amdgcn_wmma_f32_16x16x32_bf16(false, Ay.v, false, wB.v,
                                                  (short)0, acc, false, false);
#pragma unroll
    for (int q = 0; q < 4; q++) {
      v2f av = { hA[2 * q], hA[2 * q + 1] };
      v2f bv = { bB[2 * q], bB[2 * q + 1] };
      acc = __builtin_amdgcn_wmma_f32_16x16x4_f32(false, av, false, bv,
                                                  (short)0, acc, false, false);
    }

    float act[8];
    if (w == 2) {
#pragma unroll
      for (int i = 0; i < 8; i++) act[i] = fast_tanh(acc[i]);
    } else {
#pragma unroll
      for (int i = 0; i < 8; i++) act[i] = fast_sig(acc[i]);
    }
    float* gp = &gbuf[p][w][0];
#pragma unroll
    for (int v = 0; v < 8; v++) gp[(v + 8 * hi) * 18 + n] = act[v];
    ybuf[1 - p][tid] = stage;
    __syncthreads();

    const float* g0 = &gbuf[p][0][0];
    const float* g1 = &gbuf[p][1][0];
    const float* g2 = &gbuf[p][2][0];
    const float* g3 = &gbuf[p][3][0];
#pragma unroll
    for (int q = 0; q < 4; q++) {
#pragma unroll
      for (int r = 0; r < 2; r++) {
        const int off = n * 18 + 4 * q + r + 2 * hi;
        float c = g1[off] * cA[2 * q + r] + g0[off] * g2[off];
        cA[2 * q + r] = c;
        hA[2 * q + r] = g3[off] * fast_tanh(c);
      }
    }
  }

  // final hidden state -> out[b][ch*32 + dir*16 + k], b = mt*16 + n
  if (w == 0) {
    float* op = out + (size_t)(mt * 16 + n) * (NCH * 32) + ch * 32 + dir * 16;
#pragma unroll
    for (int q = 0; q < 4; q++) {
      op[4 * q + 2 * hi]     = hA[2 * q];
      op[4 * q + 2 * hi + 1] = hA[2 * q + 1];
    }
  }
}

extern "C" void kernel_launch(void* const* d_in, const int* in_sizes, int n_in,
                              void* d_out, int out_size, void* d_ws, size_t ws_size,
                              hipStream_t stream) {
  (void)in_sizes; (void)n_in; (void)out_size; (void)ws_size;
  const float* data = (const float*)d_in[0];
  const float* Wih0 = (const float*)d_in[1];
  const float* Whh0 = (const float*)d_in[2];
  const float* bih0 = (const float*)d_in[3];
  const float* bhh0 = (const float*)d_in[4];
  const float* Wih1 = (const float*)d_in[5];
  const float* Whh1 = (const float*)d_in[6];
  const float* bih1 = (const float*)d_in[7];
  const float* bhh1 = (const float*)d_in[8];

  // workspace: layer-0 output y as bf16, [C][mtile][T][16m][32f] = 201.3 MB
  unsigned* y = (unsigned*)d_ws;

  dim3 grid(NCH * 2 * 16), block(128);
  hipLaunchKernelGGL(lstm_l0, grid, block, 0, stream,
                     data, Wih0, Whh0, bih0, bhh0, y);
  hipLaunchKernelGGL(lstm_l1, grid, block, 0, stream,
                     y, Wih1, Whh1, bih1, bhh1, (float*)d_out);
}
